// MagNet_43611097924233
// MI455X (gfx1250) — matrix operand order repaired and checked by reading, not compiled, not run on hardware
//
#include <hip/hip_runtime.h>
#include <hip/hip_bf16.h>

// ---------------- problem constants ----------------
#define NN     50000
#define EE     800000
#define EE2    1600000          // 2E symmetrized edges
#define FF     128              // IN == HID == 128 (both layers' SpMV feature width)
#define KK3    384              // (K+1)*F = 3*128 GEMM reduction depth
#define OEMB   64
#define TWO_PI_Q 1.5707963267948966f

typedef unsigned int u32;
typedef unsigned short u16;
typedef __attribute__((ext_vector_type(4)))  u32    u32x4;
typedef __attribute__((ext_vector_type(8)))  float  v8f;
typedef __attribute__((ext_vector_type(16))) __bf16 bf16x16;

union Frag { u32x4 u[2]; bf16x16 v; };

__device__ __forceinline__ u16 f2bf(float f) {           // RNE f32 -> bf16
    u32 u = __float_as_uint(f);
    u += 0x7fffu + ((u >> 16) & 1u);
    return (u16)(u >> 16);
}

// ---------------- preprocessing ----------------
__global__ void k_zero_u32(u32* p, int n) {
    int i = blockIdx.x * 256 + threadIdx.x;
    if (i < n) p[i] = 0u;
}

__global__ void k_count(const int* __restrict__ src, const int* __restrict__ dst,
                        u32* __restrict__ cnt) {
    int e = blockIdx.x * 256 + threadIdx.x;
    if (e >= EE2) return;
    int r = (e < EE) ? src[e] : dst[e - EE];
    atomicAdd(&cnt[r], 1u);
}

__global__ void k_dinv(const u32* __restrict__ cnt, float* __restrict__ dinv) {
    int i = blockIdx.x * 256 + threadIdx.x;
    if (i >= NN) return;
    float d = 0.5f * (float)cnt[i];
    dinv[i] = (d > 0.0f) ? rsqrtf(fmaxf(d, 1e-12f)) : 0.0f;
}

// two-level exclusive scan of cnt[N] -> rowptr[N+1]
__global__ void k_scan_blocks(const u32* __restrict__ cnt, u32* __restrict__ scanbuf,
                              u32* __restrict__ bsum, int n) {
    __shared__ u32 s[256];
    int t = threadIdx.x, i = blockIdx.x * 256 + t;
    s[t] = (i < n) ? cnt[i] : 0u;
    __syncthreads();
    for (int d = 1; d < 256; d <<= 1) {
        u32 add = (t >= d) ? s[t - d] : 0u;
        __syncthreads();
        s[t] += add;
        __syncthreads();
    }
    if (i < n) scanbuf[i] = s[t];
    if (t == 255) bsum[blockIdx.x] = s[255];
}

__global__ void k_scan_partials(u32* __restrict__ bsum, int nb) {
    __shared__ u32 s[256];
    int t = threadIdx.x;
    s[t] = (t < nb) ? bsum[t] : 0u;
    __syncthreads();
    for (int d = 1; d < 256; d <<= 1) {
        u32 add = (t >= d) ? s[t - d] : 0u;
        __syncthreads();
        s[t] += add;
        __syncthreads();
    }
    if (t < nb) bsum[t] = s[t];
}

__global__ void k_finalize(const u32* __restrict__ cnt, const u32* __restrict__ scanbuf,
                           const u32* __restrict__ bsum, u32* __restrict__ rowptr,
                           u32* __restrict__ cursor, int n) {
    int b = blockIdx.x, t = threadIdx.x, i = b * 256 + t;
    if (i >= n) return;
    u32 off  = (b > 0) ? bsum[b - 1] : 0u;
    u32 incl = scanbuf[i] + off;
    u32 excl = incl - cnt[i];
    rowptr[i] = excl;
    cursor[i] = excl;
    if (i == n - 1) rowptr[n] = incl;
}

__global__ void k_scatter(const int* __restrict__ src, const int* __restrict__ dst,
                          const float* __restrict__ dinv, u32* __restrict__ cursor,
                          int* __restrict__ ccol, float* __restrict__ cvr,
                          float* __restrict__ cvi) {
    int e = blockIdx.x * 256 + threadIdx.x;
    if (e >= EE2) return;
    int r, c; float s;
    if (e < EE) { r = src[e]; c = dst[e]; s = 1.0f; }
    else        { r = dst[e - EE]; c = src[e - EE]; s = -1.0f; }
    float nrm = dinv[r] * 0.5f * dinv[c];
    float ct = cosf(TWO_PI_Q);           // cos is even in theta
    float st = s * sinf(TWO_PI_Q);       // sin is odd
    u32 p = atomicAdd(&cursor[r], 1u);
    ccol[p] = c;
    cvr[p] = -nrm * ct;
    cvi[p] = -nrm * st;
}

// deterministic ordering (and gather locality): insertion-sort each row by (col, sign)
__global__ void k_sortrows(const u32* __restrict__ rowptr, int* __restrict__ ccol,
                           float* __restrict__ cvr, float* __restrict__ cvi) {
    int r = blockIdx.x * 256 + threadIdx.x;
    if (r >= NN) return;
    u32 beg = rowptr[r], end = rowptr[r + 1];
    for (u32 j = beg + 1; j < end; ++j) {
        int c = ccol[j]; float a = cvr[j], b = cvi[j];
        u32 key = ((u32)c << 1) | (__float_as_uint(b) >> 31);
        u32 k = j;
        while (k > beg) {
            int   cp = ccol[k - 1];
            float bp = cvi[k - 1];
            u32 kp = ((u32)cp << 1) | (__float_as_uint(bp) >> 31);
            if (kp <= key) break;
            ccol[k] = cp; cvr[k] = cvr[k - 1]; cvi[k] = bp;
            --k;
        }
        ccol[k] = c; cvr[k] = a; cvi[k] = b;
    }
}

// W [3,F,O] f32  ->  Wt [O, 3F] bf16 (transposed so B-fragments load contiguous in K)
__global__ void k_convW(const float* __restrict__ W, u16* __restrict__ Wt, int O, int tot) {
    int i = blockIdx.x * 256 + threadIdx.x;
    if (i >= tot) return;
    int o = i % O, kf = i / O;
    Wt[(size_t)o * KK3 + kf] = f2bf(W[i]);
}

// ---------------- sparse propagation (CSR, gather-only, no atomics) ----------------
__global__ __launch_bounds__(FF)
void k_spmv_dual(const float* __restrict__ x, const u32* __restrict__ rowptr,
                 const int* __restrict__ ccol, const float* __restrict__ va,
                 const float* __restrict__ vb, float* __restrict__ oa,
                 float* __restrict__ ob) {
    int row = blockIdx.x, f = threadIdx.x;
    u32 beg = rowptr[row], end = rowptr[row + 1];
    float ar = 0.0f, ai = 0.0f;
    for (u32 j = beg; j < end; ++j) {
        int c = ccol[j];
        if (j + 1 < end) __builtin_prefetch(&x[(size_t)ccol[j + 1] * FF + f], 0, 1);
        float xv = x[(size_t)c * FF + f];
        ar = fmaf(va[j], xv, ar);
        ai = fmaf(vb[j], xv, ai);
    }
    oa[(size_t)row * FF + f] = ar;
    ob[(size_t)row * FF + f] = ai;
}

__global__ __launch_bounds__(FF)
void k_spmv_one(const float* __restrict__ x, const u32* __restrict__ rowptr,
                const int* __restrict__ ccol, const float* __restrict__ val,
                float* __restrict__ out) {
    int row = blockIdx.x, f = threadIdx.x;
    u32 beg = rowptr[row], end = rowptr[row + 1];
    float acc = 0.0f;
    for (u32 j = beg; j < end; ++j) {
        int c = ccol[j];
        if (j + 1 < end) __builtin_prefetch(&x[(size_t)ccol[j + 1] * FF + f], 0, 1);
        acc = fmaf(val[j], x[(size_t)c * FF + f], acc);
    }
    out[(size_t)row * FF + f] = acc;
}

// linear combine of up to 4 f32 stream buffers -> bf16 slice of U [N, 384]
__global__ void k_combine(u16* __restrict__ U, int slice,
                          const float* p0, float a0, const float* p1, float a1,
                          const float* p2, float a2, const float* p3, float a3) {
    int i = blockIdx.x * 256 + threadIdx.x;
    if (i >= NN * FF) return;
    int n = i >> 7, f = i & 127;
    float v = 0.0f;
    if (p0) v = fmaf(a0, p0[i], v);
    if (p1) v = fmaf(a1, p1[i], v);
    if (p2) v = fmaf(a2, p2[i], v);
    if (p3) v = fmaf(a3, p3[i], v);
    U[(size_t)n * KK3 + slice * FF + f] = f2bf(v);
}

// ---------------- dense combine: bf16 WMMA GEMM + bias + complex ReLU ----------------
// D = U * Wt^T : per wave a 16x16 tile, K=384 in 12 steps of v_wmma_f32_16x16x32_bf16.
__global__ __launch_bounds__(128)
void k_gemm_wmma(const u16* __restrict__ Ur, const u16* __restrict__ Ui,
                 const u16* __restrict__ Wt, const float* __restrict__ bias,
                 float* __restrict__ xr_out, float* __restrict__ xi_out, int O) {
    int lane = threadIdx.x & 31;
    int wave = threadIdx.x >> 5;
    int rt = blockIdx.x * 4 + wave;
    if (rt >= NN / 16) return;                 // 3125 row tiles
    int ct = blockIdx.y;
    int m  = lane & 15;                        // A row within tile / D column
    int kg = lane >> 4;                        // lane-group (K-half select)
    const u16* arow = Ur + (size_t)(rt * 16 + m) * KK3;
    const u16* irow = Ui + (size_t)(rt * 16 + m) * KK3;
    const u16* brow = Wt + (size_t)(ct * 16 + m) * KK3;   // column m of B, contiguous K
    v8f accr = {0.f,0.f,0.f,0.f,0.f,0.f,0.f,0.f};
    v8f acci = {0.f,0.f,0.f,0.f,0.f,0.f,0.f,0.f};
#pragma unroll 4
    for (int kk = 0; kk < KK3; kk += 32) {
        Frag fa, fi, fb;
        // A (16x32 bf16): lane-group 0 holds K[0..7]+K[16..23], group 1 holds K[8..15]+K[24..31]
        fa.u[0] = *(const u32x4*)(arow + kk + kg * 8);
        fa.u[1] = *(const u32x4*)(arow + kk + 16 + kg * 8);
        fi.u[0] = *(const u32x4*)(irow + kk + kg * 8);
        fi.u[1] = *(const u32x4*)(irow + kk + 16 + kg * 8);
        // B (32x16 bf16): lane-group 0 holds K[0..7]+K[16..23] of col m, group 1 holds K[8..15]+K[24..31]
        fb.u[0] = *(const u32x4*)(brow + kk + kg * 8);
        fb.u[1] = *(const u32x4*)(brow + kk + 16 + kg * 8);
        accr = __builtin_amdgcn_wmma_f32_16x16x32_bf16(false, fa.v, false, fb.v,
                                                       (short)0, accr, false, false);
        acci = __builtin_amdgcn_wmma_f32_16x16x32_bf16(false, fi.v, false, fb.v,
                                                       (short)0, acci, false, false);
    }
    float bcol = bias[ct * 16 + m];
#pragma unroll
    for (int v = 0; v < 8; ++v) {
        int row = rt * 16 + v + kg * 8;        // D: VGPR v holds M = v (+8 for lanes 16-31)
        float re = accr[v] + bcol;
        float im = acci[v] + bcol;
        float msk = (re >= 0.0f) ? 1.0f : 0.0f;
        size_t idx = (size_t)row * O + ct * 16 + m;
        xr_out[idx] = re * msk;
        xi_out[idx] = im * msk;
    }
}

// ---------------- orchestration ----------------
extern "C" void kernel_launch(void* const* d_in, const int* in_sizes, int n_in,
                              void* d_out, int out_size, void* d_ws, size_t ws_size,
                              hipStream_t stream) {
    const float* x   = (const float*)d_in[0];
    const int*   src = (const int*)d_in[1];            // edge_index[0]
    const int*   dst = (const int*)d_in[1] + EE;       // edge_index[1]
    const float* W1  = (const float*)d_in[2];
    const float* b1  = (const float*)d_in[3];
    const float* W2  = (const float*)d_in[4];
    const float* b2  = (const float*)d_in[5];
    float* out_r = (float*)d_out;                      // [N, EMB]
    float* out_i = (float*)d_out + (size_t)NN * OEMB;  // [N, EMB]

    // bump allocator over d_ws (~355 MB total)
    char* base = (char*)d_ws;
    size_t off = 0;
    auto alloc = [&](size_t bytes) -> void* {
        void* p = base + off;
        off += (bytes + 255) & ~(size_t)255;
        return p;
    };
    u32*   cnt     = (u32*)alloc((size_t)NN * 4);
    u32*   rowptr  = (u32*)alloc((size_t)(NN + 1) * 4);
    u32*   cursor  = (u32*)alloc((size_t)NN * 4);      // scanbuf, then scatter cursor
    u32*   bsum    = (u32*)alloc(256 * 4);
    float* dinv    = (float*)alloc((size_t)NN * 4);
    int*   ccol    = (int*)alloc((size_t)EE2 * 4);
    float* cvr     = (float*)alloc((size_t)EE2 * 4);
    float* cvi     = (float*)alloc((size_t)EE2 * 4);
    float* S[8];
    for (int i = 0; i < 8; ++i) S[i] = (float*)alloc((size_t)NN * FF * 4);
    u16*   Ur      = (u16*)alloc((size_t)NN * KK3 * 2);
    u16*   Ui      = (u16*)alloc((size_t)NN * KK3 * 2);
    float* xr1     = (float*)alloc((size_t)NN * FF * 4);
    float* xi1     = (float*)alloc((size_t)NN * FF * 4);
    u16*   Wt1     = (u16*)alloc((size_t)FF * KK3 * 2);
    u16*   Wt2     = (u16*)alloc((size_t)OEMB * KK3 * 2);

    const int NB   = (NN + 255) / 256;                 // 196
    const int EB   = (EE2 + 255) / 256;                // 6250
    const int CB   = (NN * FF + 255) / 256;            // 25000
    const int GRX  = (NN / 16 + 3) / 4;                // 782 (4 waves/block)

    // ---- CSR build + edge values ----
    k_zero_u32     <<<NB, 256, 0, stream>>>(cnt, NN);
    k_count        <<<EB, 256, 0, stream>>>(src, dst, cnt);
    k_dinv         <<<NB, 256, 0, stream>>>(cnt, dinv);
    k_scan_blocks  <<<NB, 256, 0, stream>>>(cnt, cursor, bsum, NN);
    k_scan_partials<<<1,  256, 0, stream>>>(bsum, NB);
    k_finalize     <<<NB, 256, 0, stream>>>(cnt, cursor, bsum, rowptr, cursor, NN);
    k_scatter      <<<EB, 256, 0, stream>>>(src, dst, dinv, cursor, ccol, cvr, cvi);
    k_sortrows     <<<NB, 256, 0, stream>>>(rowptr, ccol, cvr, cvi);

    // ---- weights -> transposed bf16 ----
    k_convW<<<(3 * FF * FF  + 255) / 256, 256, 0, stream>>>(W1, Wt1, FF,   3 * FF * FF);
    k_convW<<<(3 * FF * OEMB + 255) / 256, 256, 0, stream>>>(W2, Wt2, OEMB, 3 * FF * OEMB);

    // ---- layer 1 (xr == xi == x: only 2 distinct Chebyshev streams, 4 SpMVs) ----
    k_spmv_dual<<<NN, FF, 0, stream>>>(x, rowptr, ccol, cvr, cvi, S[0], S[1]); // P(x,vr), P(x,vi)
    k_spmv_one <<<NN, FF, 0, stream>>>(S[0], rowptr, ccol, cvr, S[2]);         // P(P(x,vr),vr)
    k_spmv_one <<<NN, FF, 0, stream>>>(S[1], rowptr, ccol, cvi, S[3]);         // P(P(x,vi),vi)
    // Ur = Trr - Tii : slice0 = 0, slice1 = S0-S1, slice2 = 2*S2 - 2*S3
    k_combine<<<CB, 256, 0, stream>>>(Ur, 0, nullptr,0.f, nullptr,0.f, nullptr,0.f, nullptr,0.f);
    k_combine<<<CB, 256, 0, stream>>>(Ur, 1, S[0],1.f, S[1],-1.f, nullptr,0.f, nullptr,0.f);
    k_combine<<<CB, 256, 0, stream>>>(Ur, 2, S[2],2.f, S[3],-2.f, nullptr,0.f, nullptr,0.f);
    // Ui = Tir + Tri : slice0 = 2x, slice1 = S0+S1, slice2 = 2*S2 + 2*S3 - 2x
    k_combine<<<CB, 256, 0, stream>>>(Ui, 0, x,2.f, nullptr,0.f, nullptr,0.f, nullptr,0.f);
    k_combine<<<CB, 256, 0, stream>>>(Ui, 1, S[0],1.f, S[1],1.f, nullptr,0.f, nullptr,0.f);
    k_combine<<<CB, 256, 0, stream>>>(Ui, 2, S[2],2.f, S[3],2.f, x,-2.f, nullptr,0.f);
    k_gemm_wmma<<<dim3(GRX, FF / 16), 128, 0, stream>>>(Ur, Ui, Wt1, b1, xr1, xi1, FF);

    // ---- layer 2 (4 streams, 8 SpMVs; duals share the gathered input row) ----
    k_spmv_dual<<<NN, FF, 0, stream>>>(xr1, rowptr, ccol, cvr, cvi, S[0], S[4]); // P(xr,vr), P(xr,vi)
    k_spmv_dual<<<NN, FF, 0, stream>>>(xi1, rowptr, ccol, cvi, cvr, S[1], S[5]); // P(xi,vi), P(xi,vr)
    k_spmv_one <<<NN, FF, 0, stream>>>(S[0], rowptr, ccol, cvr, S[2]);
    k_spmv_one <<<NN, FF, 0, stream>>>(S[1], rowptr, ccol, cvi, S[3]);
    k_spmv_one <<<NN, FF, 0, stream>>>(S[4], rowptr, ccol, cvi, S[6]);
    k_spmv_one <<<NN, FF, 0, stream>>>(S[5], rowptr, ccol, cvr, S[7]);
    // Ur = Trr - Tii
    k_combine<<<CB, 256, 0, stream>>>(Ur, 0, xr1,1.f, xi1,-1.f, nullptr,0.f, nullptr,0.f);
    k_combine<<<CB, 256, 0, stream>>>(Ur, 1, S[0],1.f, S[1],-1.f, nullptr,0.f, nullptr,0.f);
    k_combine<<<CB, 256, 0, stream>>>(Ur, 2, S[2],2.f, S[3],-2.f, xr1,-1.f, xi1,1.f);
    // Ui = Tir + Tri
    k_combine<<<CB, 256, 0, stream>>>(Ui, 0, xr1,1.f, xi1,1.f, nullptr,0.f, nullptr,0.f);
    k_combine<<<CB, 256, 0, stream>>>(Ui, 1, S[4],1.f, S[5],1.f, nullptr,0.f, nullptr,0.f);
    k_combine<<<CB, 256, 0, stream>>>(Ui, 2, S[6],2.f, S[7],2.f, xr1,-1.f, xi1,-1.f);
    k_gemm_wmma<<<dim3(GRX, OEMB / 16), 128, 0, stream>>>(Ur, Ui, Wt2, b2, out_r, out_i, OEMB);
}
